// SparseMLPWithLoRA_35837207118657
// MI455X (gfx1250) — matrix operand order repaired and checked by reading, not compiled, not run on hardware
//
#include <hip/hip_runtime.h>
#include <stdint.h>

// Problem constants (B=4, S=2048, H=2048, E=8, F=H/E=256, TOPK=2, R=16)
#define TOK 8192
#define HD  2048
#define NE  8
#define FF  256
#define LR  16

typedef __attribute__((ext_vector_type(16))) __bf16 v16bf;
typedef __attribute__((ext_vector_type(8)))  float  v8f;
typedef __attribute__((ext_vector_type(4)))  unsigned int v4u;
typedef __attribute__((ext_vector_type(8)))  int v8i_t;
typedef __attribute__((ext_vector_type(4)))  int v4i_t;

union BFrag {
  uint4 q[2];
  v16bf v;
};

#define WMMA_BF16(A, B, C) \
  __builtin_amdgcn_wmma_f32_16x16x32_bf16(false, (A), false, (B), (short)0, (C), false, false)

#if defined(__has_builtin)
#if __has_builtin(__builtin_amdgcn_tensor_load_to_lds) && __has_builtin(__builtin_amdgcn_s_wait_tensorcnt)
#define HAVE_TDM 1
#endif
#endif
#ifndef HAVE_TDM
#define HAVE_TDM 0
#endif

__device__ __forceinline__ uint16_t f2bf(float f) {
  uint32_t u = __float_as_uint(f);
  u += 0x7FFFu + ((u >> 16) & 1u);   // round-to-nearest-even
  return (uint16_t)(u >> 16);
}

// ---------------------------------------------------------------------------
// fp32 -> bf16 conversion, 4 elements/thread (b128 load, b64 store)
// ---------------------------------------------------------------------------
__global__ void convert_bf16_kernel(const float* __restrict__ in,
                                    uint16_t* __restrict__ out, int n4) {
  int i = blockIdx.x * blockDim.x + threadIdx.x;
  if (i < n4) {
    float4 v = ((const float4*)in)[i];
    union { uint16_t h[4]; uint2 u; } r;
    r.h[0] = f2bf(v.x); r.h[1] = f2bf(v.y);
    r.h[2] = f2bf(v.z); r.h[3] = f2bf(v.w);
    ((uint2*)out)[i] = r.u;
  }
}

// ---------------------------------------------------------------------------
// Router: scores = softmax(x @ G), top-2, dense per-expert weight map [T,E]
// One wave per token.
// ---------------------------------------------------------------------------
__global__ __launch_bounds__(256) void router_kernel(
    const float* __restrict__ x, const float* __restrict__ G,
    float* __restrict__ wrt) {
  const int wave = (blockIdx.x * blockDim.x + threadIdx.x) >> 5;
  const int lane = threadIdx.x & 31;
  if (wave >= TOK) return;
  const float* xr = x + (size_t)wave * HD;

  float acc[NE];
#pragma unroll
  for (int e = 0; e < NE; ++e) acc[e] = 0.0f;

  for (int h = lane; h < HD; h += 32) {
    float xv = xr[h];
    const float4* g4 = (const float4*)(G + (size_t)h * NE);
    float4 g0 = g4[0], g1 = g4[1];
    acc[0] += xv * g0.x; acc[1] += xv * g0.y;
    acc[2] += xv * g0.z; acc[3] += xv * g0.w;
    acc[4] += xv * g1.x; acc[5] += xv * g1.y;
    acc[6] += xv * g1.z; acc[7] += xv * g1.w;
  }
#pragma unroll
  for (int e = 0; e < NE; ++e) {
#pragma unroll
    for (int s = 16; s > 0; s >>= 1) acc[e] += __shfl_xor(acc[e], s, 32);
  }
  if (lane == 0) {
    float m = acc[0];
#pragma unroll
    for (int e = 1; e < NE; ++e) m = fmaxf(m, acc[e]);
    float p[NE];
    float sum = 0.0f;
#pragma unroll
    for (int e = 0; e < NE; ++e) { p[e] = __expf(acc[e] - m); sum += p[e]; }
    float inv = 1.0f / sum;
#pragma unroll
    for (int e = 0; e < NE; ++e) p[e] *= inv;
    // top-2 (first index wins ties, matching jax top_k)
    int i1 = 0;
#pragma unroll
    for (int e = 1; e < NE; ++e) if (p[e] > p[i1]) i1 = e;
    int i2 = (i1 == 0) ? 1 : 0;
#pragma unroll
    for (int e = 0; e < NE; ++e) if (e != i1 && p[e] > p[i2]) i2 = e;
    float* wo = wrt + (size_t)wave * NE;
#pragma unroll
    for (int e = 0; e < NE; ++e)
      wo[e] = (e == i1) ? p[i1] : ((e == i2) ? p[i2] : 0.0f);
  }
}

// ---------------------------------------------------------------------------
// Per-expert gate/up GEMM + silu + gating + router weight -> gated[E,T,F] bf16
// grid (T/32, E), block 256 = 8 waves. Wave: 1 M-tile (16 tok) x 4 N-tiles.
// x tile (32 tok x 64 K) staged to LDS via TDM (tensor_load_to_lds) with
// padded rows (72-half stride); weights streamed per-lane as B fragments.
// ---------------------------------------------------------------------------
#define XS_STRIDE 72  // 64 K halves + 8 pad halves (TDM pad: 4 DW / 32 DW row)

__global__ __launch_bounds__(256) void moe_gateup_kernel(
    const uint16_t* __restrict__ xb, const uint16_t* __restrict__ wg,
    const uint16_t* __restrict__ wu, const float* __restrict__ wrt,
    uint16_t* __restrict__ gated) {
  __shared__ uint16_t xs[32 * XS_STRIDE];

  const int e = blockIdx.y;
  const int t0 = blockIdx.x * 32;
  const int tid = threadIdx.x;
  const int wave = tid >> 5;
  const int lane = tid & 31;
  const int mt = wave >> 2;  // token sub-tile 0..1
  const int ng = wave & 3;   // N group 0..3
  const int lhalf = (lane < 16) ? 0 : 1;
  const int lrow = lane & 15;

  v8f accg[4], accu[4];
#pragma unroll
  for (int j = 0; j < 4; ++j) { accg[j] = {}; accu[j] = {}; }

  const size_t wbase = (size_t)e * FF * HD;
  const int koff = lhalf * 16;  // B K-chunk start within 32-wide K step

  for (int h0 = 0; h0 < HD; h0 += 64) {
    __syncthreads();  // previous tile fully consumed
#if HAVE_TDM
    if (wave == 0) {
      // Tensor DMA: 2D tile 64(x) x 32(rows) of 2-byte elems, row-padded LDS
      uint64_t ga = (uint64_t)(uintptr_t)(xb + (size_t)t0 * HD + h0);
      uint32_t lds = (uint32_t)(size_t)(&xs[0]);
      v4u g0;
      g0[0] = 1u;                                        // count=1
      g0[1] = lds;                                       // lds_addr
      g0[2] = (uint32_t)ga;                              // global_addr[31:0]
      g0[3] = ((uint32_t)(ga >> 32) & 0x01FFFFFFu) |     // global_addr[56:32]
              (2u << 30);                                // type=2 (image)
      v8i_t g1;
      g1[0] = (1 << 16)      // data_size = 2 bytes
            | (1 << 20)      // pad_enable
            | (4 << 22)      // pad_interval: 32 DWORDs (one 64-half row)
            | (3 << 25);     // pad_amount: 4 DWORDs (8 halves)
      g1[1] = (int)((unsigned)HD << 16);        // tensor_dim0[15:0] in [31:16]
      g1[2] = (int)((unsigned)TOK << 16);       // dim0 hi=0, tensor_dim1[15:0]
      g1[3] = (int)(64u << 16);                 // dim1 hi=0, tile_dim0=64
      g1[4] = 32;                               // tile_dim1=32, tile_dim2=0
      g1[5] = HD;                               // tensor_dim0_stride[31:0]
      g1[6] = 0;                                // stride hi, dim1_stride lo
      g1[7] = 0;
      v4i_t gz4 = {0, 0, 0, 0};
      v8i_t gz8 = {0, 0, 0, 0, 0, 0, 0, 0};
      __builtin_amdgcn_tensor_load_to_lds(g0, g1, gz4, gz4, gz8, 0);
      __builtin_amdgcn_s_wait_tensorcnt(0);
    }
    __syncthreads();
#else
    {
      // manual staging: 256 threads x one b128 chunk (8 halves)
      const int r = tid >> 3;
      const int c = (tid & 7) * 8;
      *(uint4*)&xs[r * XS_STRIDE + c] =
          *(const uint4*)&xb[(size_t)(t0 + r) * HD + h0 + c];
      __syncthreads();
    }
#endif

#pragma unroll
    for (int kk = 0; kk < 64; kk += 32) {
      BFrag a;
      const int ao = (mt * 16 + lrow) * XS_STRIDE + kk + lhalf * 16;
      a.q[0] = *(const uint4*)&xs[ao];
      a.q[1] = *(const uint4*)&xs[ao + 8];

      BFrag bg[4], bu[4];
#pragma unroll
      for (int j = 0; j < 4; ++j) {
        const int f = (ng * 4 + j) * 16 + lrow;
        const uint16_t* pg = wg + wbase + (size_t)f * HD + h0 + kk + koff;
        const uint16_t* pu = wu + wbase + (size_t)f * HD + h0 + kk + koff;
        bg[j].q[0] = *(const uint4*)pg;  bg[j].q[1] = *(const uint4*)(pg + 8);
        bu[j].q[0] = *(const uint4*)pu;  bu[j].q[1] = *(const uint4*)(pu + 8);
      }
#pragma unroll
      for (int j = 0; j < 4; ++j) {
        accg[j] = WMMA_BF16(a.v, bg[j].v, accg[j]);
        accu[j] = WMMA_BF16(a.v, bu[j].v, accu[j]);
      }
    }
  }

  const int rbase = t0 + mt * 16 + lhalf * 8;
  float wv[8];
#pragma unroll
  for (int r = 0; r < 8; ++r) wv[r] = wrt[(size_t)(rbase + r) * NE + e];

#pragma unroll
  for (int j = 0; j < 4; ++j) {
    const int f = (ng * 4 + j) * 16 + lrow;
#pragma unroll
    for (int r = 0; r < 8; ++r) {
      float g = accg[j][r];
      float u = accu[j][r];
      float sg = g * (1.0f / (1.0f + __expf(-g)));  // silu
      gated[((size_t)e * TOK + rbase + r) * FF + f] = f2bf(sg * u * wv[r]);
    }
  }
}

// ---------------------------------------------------------------------------
// LoRA A: lr = (x @ lora_A^T) * w -> lrw[E,T,16] bf16
// grid (T/128, E), block 256 = 8 waves, each one 16x16 tile, K = H.
// ---------------------------------------------------------------------------
__global__ __launch_bounds__(256) void moe_lora_a_kernel(
    const uint16_t* __restrict__ xb, const uint16_t* __restrict__ lA,
    const float* __restrict__ wrt, uint16_t* __restrict__ lrw) {
  const int e = blockIdx.y;
  const int wave = threadIdx.x >> 5;
  const int lane = threadIdx.x & 31;
  const int t0 = blockIdx.x * 128 + wave * 16;
  const int lhalf = (lane < 16) ? 0 : 1;
  const int lrow = lane & 15;

  v8f acc = {};
  const uint16_t* arow = xb + (size_t)(t0 + lrow) * HD + lhalf * 16;
  const uint16_t* brow = lA + ((size_t)e * LR + lrow) * HD + lhalf * 16;

  for (int h0 = 0; h0 < HD; h0 += 64) {
    BFrag a0, b0, a1, b1;
    a0.q[0] = *(const uint4*)(arow + h0);
    a0.q[1] = *(const uint4*)(arow + h0 + 8);
    b0.q[0] = *(const uint4*)(brow + h0);
    b0.q[1] = *(const uint4*)(brow + h0 + 8);
    a1.q[0] = *(const uint4*)(arow + h0 + 32);
    a1.q[1] = *(const uint4*)(arow + h0 + 40);
    b1.q[0] = *(const uint4*)(brow + h0 + 32);
    b1.q[1] = *(const uint4*)(brow + h0 + 40);
    acc = WMMA_BF16(a0.v, b0.v, acc);
    acc = WMMA_BF16(a1.v, b1.v, acc);
  }

  const int rbase = t0 + lhalf * 8;
#pragma unroll
  for (int r = 0; r < 8; ++r) {
    float w = wrt[(size_t)(rbase + r) * NE + e];
    lrw[((size_t)e * TOK + rbase + r) * LR + lrow] = f2bf(acc[r] * w);
  }
}

// ---------------------------------------------------------------------------
// Combine: out[T,H] = sum_e gated[e] @ down_w[e]^T + lrw[e] @ lora_B[e]^T
// grid (T/32, H/256), block 256 = 8 waves; wave: 1 M-tile x 4 N-tiles; f32 out.
// ---------------------------------------------------------------------------
__global__ __launch_bounds__(256) void moe_combine_kernel(
    const uint16_t* __restrict__ gated, const uint16_t* __restrict__ wd,
    const uint16_t* __restrict__ lrw, const uint16_t* __restrict__ lB,
    float* __restrict__ out) {
  const int t0 = blockIdx.x * 32;
  const int hb = blockIdx.y * 256;
  const int wave = threadIdx.x >> 5;
  const int lane = threadIdx.x & 31;
  const int mt = wave >> 2;
  const int ng = wave & 3;
  const int lhalf = (lane < 16) ? 0 : 1;
  const int lrow = lane & 15;

  v8f acc[4];
#pragma unroll
  for (int j = 0; j < 4; ++j) acc[j] = {};

  const int arow_t = t0 + mt * 16 + lrow;
  uint4 zq; zq.x = 0u; zq.y = 0u; zq.z = 0u; zq.w = 0u;

  for (int e = 0; e < NE; ++e) {
    const uint16_t* abase = gated + ((size_t)e * TOK + arow_t) * FF + lhalf * 16;
    // down projection: K = FF = 256
    for (int f0 = 0; f0 < FF; f0 += 32) {
      BFrag a;
      a.q[0] = *(const uint4*)(abase + f0);
      a.q[1] = *(const uint4*)(abase + f0 + 8);
      BFrag b[4];
#pragma unroll
      for (int j = 0; j < 4; ++j) {
        const int h = hb + (ng * 4 + j) * 16 + lrow;
        const uint16_t* pb = wd + ((size_t)e * HD + h) * FF + f0 + lhalf * 16;
        b[j].q[0] = *(const uint4*)pb;
        b[j].q[1] = *(const uint4*)(pb + 8);
        __builtin_prefetch(pb + 32, 0, 1);  // next K block of the weight row
      }
#pragma unroll
      for (int j = 0; j < 4; ++j) acc[j] = WMMA_BF16(a.v, b[j].v, acc[j]);
    }
    // LoRA B: K = 16 padded to one K=32 WMMA with zeros
    {
      BFrag a;
      const uint16_t* la = lrw + ((size_t)e * TOK + arow_t) * LR;
      a.q[0] = *(const uint4*)la;        // K 0..7 (lanes>=16: B is zero)
      a.q[1] = *(const uint4*)(la + 8);  // K 8..15
      BFrag b[4];
#pragma unroll
      for (int j = 0; j < 4; ++j) {
        const int h = hb + (ng * 4 + j) * 16 + lrow;
        if (lhalf == 0) {  // lanes 0..15 carry K = 0..15 (real rank-16 block)
          const uint16_t* pb = lB + ((size_t)e * HD + h) * LR;
          b[j].q[0] = *(const uint4*)pb;
          b[j].q[1] = *(const uint4*)(pb + 8);
        } else {           // lanes 16..31 carry K = 16..31 -> zero
          b[j].q[0] = zq; b[j].q[1] = zq;
        }
      }
#pragma unroll
      for (int j = 0; j < 4; ++j) acc[j] = WMMA_BF16(a.v, b[j].v, acc[j]);
    }
  }

  const int rbase = t0 + mt * 16 + lhalf * 8;
#pragma unroll
  for (int j = 0; j < 4; ++j) {
    const int h = hb + (ng * 4 + j) * 16 + lrow;
#pragma unroll
    for (int r = 0; r < 8; ++r)
      out[(size_t)(rbase + r) * HD + h] = acc[j][r];
  }
}

// ---------------------------------------------------------------------------
// Host-side launch
// ---------------------------------------------------------------------------
static void conv(const float* in, uint16_t* outp, int n, hipStream_t s) {
  int n4 = n / 4;
  convert_bf16_kernel<<<(n4 + 255) / 256, 256, 0, s>>>(in, outp, n4);
}

extern "C" void kernel_launch(void* const* d_in, const int* in_sizes, int n_in,
                              void* d_out, int out_size, void* d_ws, size_t ws_size,
                              hipStream_t stream) {
  (void)in_sizes; (void)n_in; (void)out_size; (void)ws_size;
  const float* x      = (const float*)d_in[0];  // [T, H]
  const float* G      = (const float*)d_in[1];  // [H, E]
  const float* gate_w = (const float*)d_in[2];  // [E, F, H]
  const float* up_w   = (const float*)d_in[3];  // [E, F, H]
  const float* down_w = (const float*)d_in[4];  // [E, H, F]
  const float* lora_A = (const float*)d_in[5];  // [E, R, H]
  const float* lora_B = (const float*)d_in[6];  // [E, H, R]
  float* out = (float*)d_out;                   // [T, H] f32

  uint8_t* ws = (uint8_t*)d_ws;
  size_t off = 0;
  auto carve = [&](size_t bytes) -> uint8_t* {
    uint8_t* p = ws + off;
    off += (bytes + 255) & ~(size_t)255;
    return p;
  };
  uint16_t* xb    = (uint16_t*)carve((size_t)TOK * HD * 2);
  uint16_t* wg    = (uint16_t*)carve((size_t)NE * FF * HD * 2);
  uint16_t* wu    = (uint16_t*)carve((size_t)NE * FF * HD * 2);
  uint16_t* wdw   = (uint16_t*)carve((size_t)NE * HD * FF * 2);
  uint16_t* lAb   = (uint16_t*)carve((size_t)NE * LR * HD * 2);
  uint16_t* lBb   = (uint16_t*)carve((size_t)NE * HD * LR * 2);
  float*    wrt   = (float*)   carve((size_t)TOK * NE * 4);
  uint16_t* gated = (uint16_t*)carve((size_t)NE * TOK * FF * 2);
  uint16_t* lrw   = (uint16_t*)carve((size_t)NE * TOK * LR * 2);

  // fp32 -> bf16 conversions
  conv(x,      xb,  TOK * HD,      stream);
  conv(gate_w, wg,  NE * FF * HD,  stream);
  conv(up_w,   wu,  NE * FF * HD,  stream);
  conv(down_w, wdw, NE * HD * FF,  stream);
  conv(lora_A, lAb, NE * LR * HD,  stream);
  conv(lora_B, lBb, NE * HD * LR,  stream);

  // Router (one wave per token, 8 waves per block)
  router_kernel<<<TOK / 8, 256, 0, stream>>>(x, G, wrt);

  // Expert gate/up + gating
  moe_gateup_kernel<<<dim3(TOK / 32, NE), 256, 0, stream>>>(xb, wg, wu, wrt, gated);

  // LoRA A projection
  moe_lora_a_kernel<<<dim3(TOK / 128, NE), 256, 0, stream>>>(xb, lAb, wrt, lrw);

  // Down projection + LoRA B + combine over experts
  moe_combine_kernel<<<dim3(TOK / 32, HD / 256), 256, 0, stream>>>(gated, wdw, lrw, lBb, out);
}
